// Self_Attention_14946486190233
// MI455X (gfx1250) — matrix operand order repaired and checked by reading, not compiled, not run on hardware
//
#include <hip/hip_runtime.h>

#define NB 8
#define NN 4096
#define DM 128

typedef __attribute__((ext_vector_type(16))) _Float16 v16h;
typedef __attribute__((ext_vector_type(8)))  _Float16 v8h;
typedef __attribute__((ext_vector_type(8)))  float    v8f;

static __device__ __forceinline__ v16h cat8(v8h lo, v8h hi) {
  return __builtin_shufflevector(lo, hi, 0,1,2,3,4,5,6,7,8,9,10,11,12,13,14,15);
}

// ---------------------------------------------------------------------------
// Kernel 0: convert the three 128x128 fp32 weight matrices to fp16.
// ---------------------------------------------------------------------------
__global__ void wcvt(const float* __restrict__ Wq, const float* __restrict__ Wk,
                     const float* __restrict__ Wv, _Float16* __restrict__ Wh) {
  int i = blockIdx.x * 256 + threadIdx.x;          // 0 .. 16383
  Wh[i]               = (_Float16)Wq[i];
  Wh[DM * DM + i]     = (_Float16)Wk[i];
  Wh[2 * DM * DM + i] = (_Float16)Wv[i];
}

// ---------------------------------------------------------------------------
// Kernel 1: QKV projection with WMMA. One wave per 16-token tile.
//   Qh[b][n][e] fp16, Vh[b][n][e] fp16 (row-major), KT[b][e][n] fp16 (transposed)
// mat-loop deliberately NOT unrolled: keeps <=8 live accumulators (vs 24) so
// the kernel stays under 256 VGPRs (no s_set_vgpr_msb) and retains occupancy.
// ---------------------------------------------------------------------------
__global__ void __launch_bounds__(256) qkv_proj(
    const float* __restrict__ x, const _Float16* __restrict__ Wh,
    const float* __restrict__ bq, const float* __restrict__ bk,
    const float* __restrict__ bv,
    _Float16* __restrict__ Qh, _Float16* __restrict__ KT, _Float16* __restrict__ Vh)
{
  const int lane = threadIdx.x & 31;
  const int wave = threadIdx.x >> 5;
  const int tile = blockIdx.x * 8 + wave;          // 0 .. 2047 (B*N/16)
  const int lrow = lane & 15;
  const int hf   = lane >> 4;

  // A-fragments of x (16x32 fp16 per chunk): lane holds row (lane&15),
  // halves K = base..base+7 and base+16..base+23, base = hf*8 + 32*c.
  v16h a[4];
#pragma unroll
  for (int c = 0; c < 4; ++c) {
    const float* p = x + (size_t)(tile * 16 + lrow) * DM + c * 32 + hf * 8;
    v16h t;
#pragma unroll
    for (int i = 0; i < 8; ++i) { t[i] = (_Float16)p[i]; t[i + 8] = (_Float16)p[i + 16]; }
    a[c] = t;
  }

  const int b  = tile >> 8;                        // batch
  const int m0 = (tile & 255) * 16;                // token base within batch
  const int r0 = m0 + hf * 8;                      // row base for this lane-half (C layout)

  const float* biases[3] = {bq, bk, bv};

#pragma unroll 1
  for (int mat = 0; mat < 3; ++mat) {
    const _Float16* W = Wh + (size_t)mat * DM * DM;
#pragma unroll
    for (int et = 0; et < 8; ++et) {
      v8f acc = {};
#pragma unroll
      for (int c = 0; c < 4; ++c) {
        // B-fragment: lane holds output col (et*16+lrow), K rows in table order.
        v16h bf;
#pragma unroll
        for (int i = 0; i < 8; ++i) {
          bf[i]     = W[(size_t)(c * 32 + hf * 8 + i) * DM + et * 16 + lrow];
          bf[i + 8] = W[(size_t)(c * 32 + 16 + hf * 8 + i) * DM + et * 16 + lrow];
        }
        acc = __builtin_amdgcn_wmma_f32_16x16x32_f16(false, a[c], false, bf,
                                                     (short)0, acc, false, false);
      }
      const int   ecol = et * 16 + lrow;
      const float bias = biases[mat][ecol];
      if (mat == 1) {
        // K stored transposed: KT[b][e][n]; 8 tokens contiguous per lane.
        _Float16* kp = KT + ((size_t)b * DM + ecol) * NN + r0;
        v8h kv;
#pragma unroll
        for (int r = 0; r < 8; ++r) kv[r] = (_Float16)(acc[r] + bias);
        *(v8h*)kp = kv;
      } else {
        _Float16* dst = (mat == 0 ? Qh : Vh) + ((size_t)b * NN + r0) * DM + ecol;
#pragma unroll
        for (int r = 0; r < 8; ++r) dst[(size_t)r * DM] = (_Float16)(acc[r] + bias);
      }
    }
  }
}

// ---------------------------------------------------------------------------
// Kernel 2: flash attention. 8 waves/block, one 16-query tile per wave, all
// waves of a block stream the SAME K/V tiles in lockstep (per-iteration
// __syncthreads) so 7/8 of the K/V reads hit the WGP cache instead of L2.
// ---------------------------------------------------------------------------
__global__ void __launch_bounds__(256) attn(
    const _Float16* __restrict__ Qh, const _Float16* __restrict__ KT,
    const _Float16* __restrict__ Vh, float* __restrict__ out)
{
  __shared__ _Float16 lds[8][16 * 32];             // per-wave P tile (16x32 fp16)
  const int lane = threadIdx.x & 31;
  const int wave = threadIdx.x >> 5;
  const int tile = blockIdx.x * 8 + wave;          // 0 .. 2047
  const int b    = tile >> 8;
  const int q0   = (tile & 255) * 16;
  const int hf   = lane >> 4;
  const int lrow = lane & 15;

  const _Float16* Qb = Qh + (size_t)b * NN * DM;
  const _Float16* Kb = KT + (size_t)b * DM * NN;
  const _Float16* Vb = Vh + (size_t)b * NN * DM;
  _Float16* myl = lds[wave];

  // Q tile as 4 A-fragments, resident in VGPRs for the whole loop.
  v16h qa[4];
#pragma unroll
  for (int c = 0; c < 4; ++c) {
    const _Float16* p = Qb + (size_t)(q0 + lrow) * DM + c * 32 + hf * 8;
    qa[c] = cat8(*(const v8h*)p, *(const v8h*)(p + 16));
  }

  v8f o[8] = {};
  float m2[8], l[8];
#pragma unroll
  for (int r = 0; r < 8; ++r) { m2[r] = -1.0e30f; l[r] = 0.0f; }

  // softmax done in exp2 domain: s2 = s * (1/sqrt(128)) * log2(e)
  const float sscale = 0.08838834764831845f * 1.44269504088896340f;

  for (int n0 = 0; n0 < NN; n0 += 32) {
    // ---- S = Q * K^T for 32 keys (two 16-col subtiles) ----
    v8f s0 = {}, s1 = {};
#pragma unroll
    for (int c = 0; c < 4; ++c) {
      v16h k0, k1;
#pragma unroll
      for (int i = 0; i < 8; ++i) {
        k0[i]     = Kb[(size_t)(c * 32 + hf * 8 + i) * NN + n0 + lrow];
        k0[i + 8] = Kb[(size_t)(c * 32 + 16 + hf * 8 + i) * NN + n0 + lrow];
        k1[i]     = Kb[(size_t)(c * 32 + hf * 8 + i) * NN + n0 + 16 + lrow];
        k1[i + 8] = Kb[(size_t)(c * 32 + 16 + hf * 8 + i) * NN + n0 + 16 + lrow];
      }
      s0 = __builtin_amdgcn_wmma_f32_16x16x32_f16(false, qa[c], false, k0,
                                                  (short)0, s0, false, false);
      s1 = __builtin_amdgcn_wmma_f32_16x16x32_f16(false, qa[c], false, k1,
                                                  (short)0, s1, false, false);
    }

    // ---- prefetch V fragments now; they fly during the softmax VALU work ----
    v16h vb[8];
#pragma unroll
    for (int et = 0; et < 8; ++et)
#pragma unroll
      for (int i = 0; i < 8; ++i) {
        vb[et][i]     = Vb[(size_t)(n0 + hf * 8 + i) * DM + et * 16 + lrow];
        vb[et][i + 8] = Vb[(size_t)(n0 + 16 + hf * 8 + i) * DM + et * 16 + lrow];
      }

    // ---- online softmax (per C-layout row r + 8*hf, cols across 16 lanes) ----
#pragma unroll
    for (int r = 0; r < 8; ++r) {
      float a0 = s0[r] * sscale, a1 = s1[r] * sscale;
      float v = fmaxf(a0, a1);
      v = fmaxf(v, __shfl_xor(v, 1, 32));
      v = fmaxf(v, __shfl_xor(v, 2, 32));
      v = fmaxf(v, __shfl_xor(v, 4, 32));
      v = fmaxf(v, __shfl_xor(v, 8, 32));
      const float nm    = fmaxf(m2[r], v);
      const float alpha = exp2f(m2[r] - nm);
      m2[r] = nm;
      a0 = exp2f(a0 - nm);
      a1 = exp2f(a1 - nm);
      float rs = a0 + a1;
      rs += __shfl_xor(rs, 1, 32);
      rs += __shfl_xor(rs, 2, 32);
      rs += __shfl_xor(rs, 4, 32);
      rs += __shfl_xor(rs, 8, 32);
      l[r] = l[r] * alpha + rs;
#pragma unroll
      for (int et = 0; et < 8; ++et) o[et][r] *= alpha;
      const int row = r + hf * 8;
      myl[row * 32 + lrow]      = (_Float16)a0;
      myl[row * 32 + 16 + lrow] = (_Float16)a1;
    }
    __syncthreads();                               // order LDS stores -> loads

    // ---- O += P * V : P re-read from LDS in A-fragment layout ----
    const _Float16* ap = myl + lrow * 32 + hf * 8;
    v16h pa = cat8(*(const v8h*)ap, *(const v8h*)(ap + 16));
    __syncthreads();                               // loads done before next overwrite
#pragma unroll
    for (int et = 0; et < 8; ++et)
      o[et] = __builtin_amdgcn_wmma_f32_16x16x32_f16(false, pa, false, vb[et],
                                                     (short)0, o[et], false, false);
  }

  // ---- normalize and write fp32 output ----
  float rinv[8];
#pragma unroll
  for (int r = 0; r < 8; ++r) rinv[r] = 1.0f / l[r];
  float* ob = out + ((size_t)b * NN + q0) * DM;
#pragma unroll
  for (int et = 0; et < 8; ++et)
#pragma unroll
    for (int r = 0; r < 8; ++r)
      ob[(size_t)(r + hf * 8) * DM + et * 16 + lrow] = o[et][r] * rinv[r];
}

// ---------------------------------------------------------------------------
extern "C" void kernel_launch(void* const* d_in, const int* in_sizes, int n_in,
                              void* d_out, int out_size, void* d_ws, size_t ws_size,
                              hipStream_t stream) {
  (void)in_sizes; (void)n_in; (void)out_size; (void)ws_size;
  const float* x  = (const float*)d_in[0];
  const float* Wq = (const float*)d_in[1];
  const float* bq = (const float*)d_in[2];
  const float* Wk = (const float*)d_in[3];
  const float* bk = (const float*)d_in[4];
  const float* Wv = (const float*)d_in[5];
  const float* bv = (const float*)d_in[6];

  char* ws = (char*)d_ws;
  const size_t QKV_BYTES = (size_t)NB * NN * DM * sizeof(_Float16); // 8 MB each
  _Float16* Wh = (_Float16*)ws;                                     // 96 KB
  _Float16* Qh = (_Float16*)(ws + (size_t)128 * 1024);
  _Float16* KT = (_Float16*)(ws + (size_t)128 * 1024 + QKV_BYTES);
  _Float16* Vh = (_Float16*)(ws + (size_t)128 * 1024 + 2 * QKV_BYTES);

  wcvt<<<64, 256, 0, stream>>>(Wq, Wk, Wv, Wh);
  qkv_proj<<<256, 256, 0, stream>>>(x, Wh, bq, bk, bv, Qh, KT, Vh);
  attn<<<256, 256, 0, stream>>>(Qh, KT, Vh, (float*)d_out);
}
